// NeuroGraphTSC_13941463843041
// MI455X (gfx1250) — matrix operand order repaired and checked by reading, not compiled, not run on hardware
//
#include <hip/hip_runtime.h>
#include <hip/hip_bf16.h>
#include <math.h>

typedef __attribute__((ext_vector_type(16))) _Float16 v16h;
typedef __attribute__((ext_vector_type(8)))  float    v8f;

#define N_NODES 32
#define GAT_F   64
#define LSTM_H  128
#define SEQ_T   512
#define BATCH   32
#define NDIM    512   // 4*H
#define LDP     168   // Abuf row pitch in halves (>=160)

// Branch-free fast transcendentals: v_exp_f32 / v_rcp_f32 are TRANS ops that
// co-execute with VALU -- no EXEC-mask branching on the sequential path.
__device__ __forceinline__ float fexp(float x) {
    return __builtin_amdgcn_exp2f(x * 1.4426950408889634f);
}
__device__ __forceinline__ float frcp(float x) {
    return __builtin_amdgcn_rcpf(x);
}
__device__ __forceinline__ float fsigmoid(float x) {
    return frcp(1.f + fexp(-x));
}
__device__ __forceinline__ float ftanh(float x) {
    // 1 - 2/(e^{2x}+1): exact limits at +/-inf, no NaN, no branches
    return 1.f - 2.f * frcp(fexp(2.f * x) + 1.f);
}

// ---------------------------------------------------------------------------
// Kernel 1: prep  -- Wred[n,4H] = sum_f w_gat[f] * W_ih[n*64+f, :]
//                    bconst[4H] = b_ih + b_hh + sum_{n,f} b_gat[f]*W_ih[n*64+f,:]
// ---------------------------------------------------------------------------
__global__ void prep_kernel(const float* __restrict__ W_ih,
                            const float* __restrict__ w_gat,
                            const float* __restrict__ b_gat,
                            const float* __restrict__ b_ih,
                            const float* __restrict__ b_hh,
                            float* __restrict__ Wred,
                            float* __restrict__ bconst) {
    int j = blockIdx.x * blockDim.x + threadIdx.x;   // column 0..511
    if (j >= NDIM) return;
    float bc = b_ih[j] + b_hh[j];
    for (int n = 0; n < N_NODES; ++n) {
        float s = 0.f;
        #pragma unroll 4
        for (int f = 0; f < GAT_F; ++f) {
            float w = W_ih[(n * GAT_F + f) * NDIM + j];
            s  += w_gat[f] * w;
            bc += b_gat[f] * w;
        }
        Wred[n * NDIM + j] = s;
    }
    bconst[j] = bc;
}

// ---------------------------------------------------------------------------
// Kernel 2: GAT scalar aggregation.  One wave32 per (b,t) step; lane = node.
// Output layout aggH[t][b][n] (f16) for coalesced per-step staging in LSTM.
// ---------------------------------------------------------------------------
__global__ void gat_kernel(const float* __restrict__ x_seq,
                           const int*   __restrict__ edge_index,
                           const float* __restrict__ w_gat,
                           const float* __restrict__ att_src,
                           const float* __restrict__ att_dst,
                           _Float16*    __restrict__ aggH,
                           int E) {
    __shared__ int   es[256], ed[256];
    __shared__ float cscd[2];
    const int Et  = E + N_NODES;               // + self loops
    const int tid = threadIdx.x;
    if (tid < E)            { es[tid] = edge_index[tid]; ed[tid] = edge_index[E + tid]; }
    else if (tid < Et)      { es[tid] = tid - E;         ed[tid] = tid - E; }
    if (tid == 0) {
        float cs = 0.f, cd = 0.f;
        for (int f = 0; f < GAT_F; ++f) { cs += w_gat[f] * att_src[f]; cd += w_gat[f] * att_dst[f]; }
        cscd[0] = cs; cscd[1] = cd;
    }
    __syncthreads();
    const float cs = cscd[0], cd = cscd[1];

    const int lane = tid & 31;
    const int s    = blockIdx.x * (blockDim.x >> 5) + (tid >> 5);  // (b,t) flat, b-major
    const int b    = s >> 9;
    const int t    = s & (SEQ_T - 1);

    const float x = x_seq[s * N_NODES + lane];

    // pass 1: per-destination max (self-loop guarantees >=1 edge per node)
    float mmax = -1e30f;
    for (int e = 0; e < Et; ++e) {
        float xs = __shfl(x, es[e]);
        float xd = __shfl(x, ed[e]);
        float l  = cs * xs + cd * xd;
        l = (l > 0.f) ? l : 0.2f * l;
        if (ed[e] == lane) mmax = fmaxf(mmax, l);
    }
    // pass 2: softmax-weighted aggregate of x[src]
    float num = 0.f, den = 0.f;
    for (int e = 0; e < Et; ++e) {
        float xs = __shfl(x, es[e]);
        float xd = __shfl(x, ed[e]);
        float l  = cs * xs + cd * xd;
        l = (l > 0.f) ? l : 0.2f * l;
        if (ed[e] == lane) {
            float ex = fexp(l - mmax);
            num += ex * xs;
            den += ex;
        }
    }
    aggH[(t * BATCH + b) * N_NODES + lane] = (_Float16)(num * frcp(den));
}

// ---------------------------------------------------------------------------
// Kernel 3: persistent single-workgroup LSTM over T=512 steps + final FC.
//   g[32,512] = [h | agg_t] (K=160) @ [W_hh ; Wred] (160x512) + bconst
// 16 waves; wave = (mtile 0..1, hblock 0..7); each wave computes the 4 gate
// tiles (i,f,g,o) for its 16x16 hidden block -> gate update is lane-local.
// B fragments live in VGPRs for the whole loop; c lives in registers; h
// round-trips through LDS (f16) each step; agg_t is software-prefetched one
// step ahead so the global load latency hides under the WMMA phase.
// ---------------------------------------------------------------------------
__global__ void __launch_bounds__(512)
lstm_kernel(const _Float16* __restrict__ aggH,
            const float*    __restrict__ W_hh,
            const float*    __restrict__ Wred,
            const float*    __restrict__ bconst,
            const float*    __restrict__ W_fc,
            const float*    __restrict__ b_fc,
            float*          __restrict__ out) {
    __shared__ _Float16 Abuf[BATCH][LDP];   // [row b][k]: k<128 = h, k in 128..159 = agg_t

    const int tid    = threadIdx.x;
    const int lane   = tid & 31;
    const int wave   = tid >> 5;
    const int mtile  = wave >> 3;     // 0..1  -> batch rows [mtile*16, +16)
    const int hblock = wave & 7;      // 0..7  -> hidden cols [hblock*16, +16)
    const int lrow   = lane & 15;
    const int lhi    = lane >> 4;

    // ---- load B fragments (weights, f32->f16) once; 4 gates x 5 k-chunks ----
    // B k order used here (contiguous, same as A): lane<16 -> col=lane, VGPR j holds K=2j,2j+1 (K 0..15);
    //                     lane>=16 -> col=lane-16, K=16+2j,16+2j+1.
    v16h bfrag[4][5];
    float bc[4];
    #pragma unroll
    for (int q = 0; q < 4; ++q) {
        const int col = q * LSTM_H + hblock * 16 + lrow;
        bc[q] = bconst[col];
        #pragma unroll
        for (int kc = 0; kc < 5; ++kc) {
            const int k0 = kc * 32 + (lhi ? 16 : 0);
            v16h f;
            #pragma unroll
            for (int j = 0; j < 8; ++j) {
                const int k = k0 + 2 * j;
                float a0, a1;
                if (k < LSTM_H) { a0 = W_hh[k * NDIM + col];            a1 = W_hh[(k + 1) * NDIM + col]; }
                else            { a0 = Wred[(k - LSTM_H) * NDIM + col]; a1 = Wred[(k - LSTM_H + 1) * NDIM + col]; }
                f[2 * j]     = (_Float16)a0;
                f[2 * j + 1] = (_Float16)a1;
            }
            bfrag[q][kc] = f;
        }
    }

    // ---- h := 0 ----
    for (int i = tid; i < BATCH * LSTM_H; i += 512)
        Abuf[i >> 7][i & 127] = (_Float16)0.f;

    float creg[8];
    #pragma unroll
    for (int r = 0; r < 8; ++r) creg[r] = 0.f;

    // agg_t prefetch pipeline: each thread owns 2 consecutive halves (one u32)
    const int sm = (tid * 2) >> 5;          // staging row (batch)
    const int sn = (tid * 2) & 31;          // staging col (node), even -> 4B aligned
    unsigned int agg_pre = *(const unsigned int*)(aggH + tid * 2);   // t = 0

    for (int t = 0; t < SEQ_T; ++t) {
        __syncthreads();                       // prev step fully consumed/produced
        // stage the prefetched agg_t, then issue prefetch for t+1 (latency
        // hides under this step's WMMA + gate phase)
        *(unsigned int*)&Abuf[sm][LSTM_H + sn] = agg_pre;
        if (t + 1 < SEQ_T)
            agg_pre = *(const unsigned int*)(aggH + (t + 1) * (BATCH * N_NODES) + tid * 2);
        __syncthreads();                       // A operand (h + agg) visible

        v8f acc0, acc1, acc2, acc3;
        #pragma unroll
        for (int r = 0; r < 8; ++r) { acc0[r] = bc[0]; acc1[r] = bc[1]; acc2[r] = bc[2]; acc3[r] = bc[3]; }

        #pragma unroll
        for (int kc = 0; kc < 5; ++kc) {
            // A k order used here (contiguous, same as B): lane<16 -> row=lane, K=k0..k0+15
            //                     lane>=16 -> row=lane-16, K=k0+16..k0+31
            const int k0 = kc * 32 + (lhi ? 16 : 0);
            const int m  = mtile * 16 + lrow;
            v16h a;
            #pragma unroll
            for (int j = 0; j < 8; ++j) {
                const int k = k0 + 2 * j;
                a[2 * j]     = Abuf[m][k];
                a[2 * j + 1] = Abuf[m][k + 1];
            }
            acc0 = __builtin_amdgcn_wmma_f32_16x16x32_f16(false, a, false, bfrag[0][kc], (short)0, acc0, false, false);
            acc1 = __builtin_amdgcn_wmma_f32_16x16x32_f16(false, a, false, bfrag[1][kc], (short)0, acc1, false, false);
            acc2 = __builtin_amdgcn_wmma_f32_16x16x32_f16(false, a, false, bfrag[2][kc], (short)0, acc2, false, false);
            acc3 = __builtin_amdgcn_wmma_f32_16x16x32_f16(false, a, false, bfrag[3][kc], (short)0, acc3, false, false);
        }
        __syncthreads();                       // all A reads done before h overwrite

        // gate update: C/D layout -> lane l, vgpr r : row = mtile*16 + lhi*8 + r,
        // col (hidden idx) = hblock*16 + lrow.  i,f,g,o all in-lane, same r.
        #pragma unroll
        for (int r = 0; r < 8; ++r) {
            const float ig = fsigmoid(acc0[r]);
            const float fg = fsigmoid(acc1[r]);
            const float gg = ftanh(acc2[r]);
            const float og = fsigmoid(acc3[r]);
            const float c  = fg * creg[r] + ig * gg;
            creg[r] = c;
            const float h = og * ftanh(c);
            const int m   = mtile * 16 + lhi * 8 + r;
            const int col = hblock * 16 + lrow;
            Abuf[m][col] = (_Float16)h;
        }
    }
    __syncthreads();

    // final classifier: out[b,cls] = h_last[b,:] @ W_fc + b_fc
    if (tid < BATCH * 4) {
        const int b = tid >> 2, cls = tid & 3;
        float s = b_fc[cls];
        for (int hd = 0; hd < LSTM_H; ++hd)
            s += (float)Abuf[b][hd] * W_fc[hd * 4 + cls];
        out[b * 4 + cls] = s;
    }
}

// ---------------------------------------------------------------------------
extern "C" void kernel_launch(void* const* d_in, const int* in_sizes, int n_in,
                              void* d_out, int out_size, void* d_ws, size_t ws_size,
                              hipStream_t stream) {
    const float* x_seq   = (const float*)d_in[0];
    const int*   edges   = (const int*)  d_in[1];
    const float* w_gat   = (const float*)d_in[2];
    const float* att_src = (const float*)d_in[3];
    const float* att_dst = (const float*)d_in[4];
    const float* b_gat   = (const float*)d_in[5];
    const float* W_ih    = (const float*)d_in[6];
    const float* W_hh    = (const float*)d_in[7];
    const float* b_ih    = (const float*)d_in[8];
    const float* b_hh    = (const float*)d_in[9];
    const float* W_fc    = (const float*)d_in[10];
    const float* b_fc    = (const float*)d_in[11];
    float* out = (float*)d_out;

    const int E = in_sizes[1] / 2;            // 96

    // workspace layout
    char* ws = (char*)d_ws;
    _Float16* aggH   = (_Float16*)ws;                                       // T*B*32 f16 = 1 MB
    float*    Wred   = (float*)(ws + (size_t)SEQ_T * BATCH * N_NODES * 2);  // 32x512 f32
    float*    bconst = Wred + N_NODES * NDIM;                               // 512 f32

    prep_kernel<<<2, 256, 0, stream>>>(W_ih, w_gat, b_gat, b_ih, b_hh, Wred, bconst);

    // 16384 (b,t) steps, 8 waves/block of 256 threads
    gat_kernel<<<(BATCH * SEQ_T) / 8, 256, 0, stream>>>(x_seq, edges, w_gat, att_src, att_dst, aggH, E);

    lstm_kernel<<<1, 512, 0, stream>>>(aggH, W_hh, Wred, bconst, W_fc, b_fc, out);
}
